// InferTree_34505767256835
// MI455X (gfx1250) — matrix operand order repaired and checked by reading, not compiled, not run on hardware
//
#include <hip/hip_runtime.h>

// ---------------- problem constants ----------------
#define B_ROWS   8192
#define DIM      2048
#define NODES    111
#define KFAN     11              // FANOUT+1
#define NCOL     1221            // NODES*KFAN logits per row
#define NTILES   80              // padded to 1280 cols (16-wide WMMA tiles)
#define KT_STEPS (DIM / 32)      // 64 k-steps of 32
#define LSTR     1236            // logits LDS row stride (floats), bank-padded
#define ASTR     2056            // A-stage LDS row stride (halves), bank-padded
#define OUT_CLASSES 1000
#define BLOCK_THREADS 512        // 16 wave32s
#define TILES_PER_WAVE 5         // 16 waves * 5 tiles = 80 tiles
#define LDS_BYTES (16 * LSTR * 4)   // 79104 B (A-stage 16*ASTR*2 = 65792 B, aliased)

typedef __attribute__((ext_vector_type(16))) _Float16 v16h;
typedef __attribute__((ext_vector_type(8)))  _Float16 v8h;
typedef __attribute__((ext_vector_type(4)))  _Float16 v4h;
typedef __attribute__((ext_vector_type(8)))  float    v8f;

// ---------------------------------------------------------------------------
// Kernel 1: repack W[node][d][k] (fp32, strided) into f16 B-fragment layout.
// For block (kt, nt): 32 lanes x 16 halves contiguous per lane;
//   lane = ((kr/8)%2)*16 + nr holds column c = nt*16+nr, K = kt*32 + (e<8 ? (lane/16)*8 + e : 16 + (lane/16)*8 + e-8).
// ---------------------------------------------------------------------------
__global__ __launch_bounds__(256) void pack_w_f16(const float* __restrict__ W,
                                                  _Float16* __restrict__ wpk) {
  long p = (long)blockIdx.x * 256 + threadIdx.x;   // packed half index (coalesced writes)
  int  e    = (int)(p & 15);
  int  lane = (int)((p >> 4) & 31);
  long blk  = p >> 9;
  int  nt   = (int)(blk % NTILES);
  int  kt   = (int)(blk / NTILES);
  int  half = lane >> 4;
  int  nr   = lane & 15;
  int  k    = kt * 32 + (e < 8 ? half * 8 + e : 16 + half * 8 + (e - 8));
  int  c    = nt * 16 + nr;
  float val = 0.f;
  if (c < NCOL) {
    int node = c / KFAN;
    int kk   = c - node * KFAN;
    val = W[((long)node * DIM + k) * KFAN + kk];   // W[node][k][kk]
  }
  wpk[p] = (_Float16)val;
}

// ---------------------------------------------------------------------------
// Kernel 2: 16 rows x 1221 cols GEMM (WMMA f16->f32) + softmax + product tree.
// 16 waves; each wave owns 5 column tiles; B fragments double-buffered so the
// loads for k-step kt+1 are in flight while the WMMAs for kt execute.
// ---------------------------------------------------------------------------
__global__ __launch_bounds__(BLOCK_THREADS) void tree_infer(
    const float* __restrict__ x,
    const float* __restrict__ bias,
    const _Float16* __restrict__ wpk,
    float* __restrict__ out) {
  extern __shared__ char smem_raw[];
  float*    lg  = (float*)smem_raw;      // phase 2: 16 x LSTR logits/probs
  _Float16* ash = (_Float16*)smem_raw;   // phase 1: 16 x ASTR f16 A-stage (aliased)

  const int  tid  = threadIdx.x;
  const int  lane = tid & 31;
  const int  w    = tid >> 5;            // wave id 0..15
  const long row0 = (long)blockIdx.x * 16;

  // ---- Stage the 16x2048 x-tile into LDS, converting fp32 -> f16 ----
  #pragma unroll
  for (int i = 0; i < 16; ++i) {
    int cf  = tid + i * BLOCK_THREADS;   // 8192 chunks of 4 floats
    int r   = cf >> 9;                   // 512 chunks per row
    int col = (cf & 511) << 2;
    const float4 v = *(const float4*)(x + (row0 + r) * (long)DIM + col);
    v4h h;
    h[0] = (_Float16)v.x; h[1] = (_Float16)v.y;
    h[2] = (_Float16)v.z; h[3] = (_Float16)v.w;
    *(v4h*)(ash + r * ASTR + col) = h;   // 8B ds_store, bank-padded stride
  }
  __syncthreads();

  // ---- Main WMMA loop ----
  v8f acc[TILES_PER_WAVE] = {};
  const int m     = lane & 15;           // A row within tile (ISA A-frag layout)
  const int khalf = lane >> 4;
  const _Float16* arow  = ash + m * ASTR + khalf * 8;
  const _Float16* bbase = wpk + (long)w * TILES_PER_WAVE * 512 + lane * 16;

  auto load_b = [&](v16h* buf, int kt) {
    const _Float16* bp = bbase + (long)kt * NTILES * 512;
    #pragma unroll
    for (int t = 0; t < TILES_PER_WAVE; ++t)
      buf[t] = *(const v16h*)(bp + t * 512);       // 32B contiguous per lane
  };
  auto load_a = [&](int kt) {
    const _Float16* ap = arow + kt * 32;
    v8h alo = *(const v8h*)(ap);                   // K = kt*32 + khalf*8 + 0..7
    v8h ahi = *(const v8h*)(ap + 16);              // K = kt*32 + 16 + khalf*8 + 0..7
    v16h a;
    #pragma unroll
    for (int i = 0; i < 8; ++i) { a[i] = alo[i]; a[i + 8] = ahi[i]; }
    return a;
  };
  auto mma = [&](v16h a, const v16h* buf) {
    #pragma unroll
    for (int t = 0; t < TILES_PER_WAVE; ++t)
      acc[t] = __builtin_amdgcn_wmma_f32_16x16x32_f16(
          /*neg_a=*/false, a, /*neg_b=*/false, buf[t],
          /*c_mod=*/(short)0, acc[t], /*reuse_a=*/false, /*reuse_b=*/false);
  };

  v16h b0[TILES_PER_WAVE], b1[TILES_PER_WAVE];
  load_b(b0, 0);
  for (int kt = 0; kt < KT_STEPS; kt += 2) {
    // step 1: compute kt from b0 while loading kt+1 into b1
    v16h a0 = load_a(kt);
    load_b(b1, kt + 1);
    __builtin_prefetch(bbase + (long)(kt + 3) * NTILES * 512, 0, 3);
    mma(a0, b0);
    // step 2: compute kt+1 from b1 while loading kt+2 into b0
    v16h a1 = load_a(kt + 1);
    int ktn = (kt + 2 < KT_STEPS) ? (kt + 2) : kt;   // clamp: harmless reload at tail
    load_b(b0, ktn);
    mma(a1, b1);
  }
  __syncthreads();   // done reading A-stage LDS; safe to alias as logits

  // ---- Spill C tiles to LDS (32-bit C/D layout: VGPR v -> row v + 8*(lane/16)) ----
  #pragma unroll
  for (int t = 0; t < TILES_PER_WAVE; ++t) {
    int nt = w * TILES_PER_WAVE + t;
    if (nt < 77) {                       // cols >= 1232 are pure padding
      int col = nt * 16 + (lane & 15);
      int rb  = (lane >> 4) * 8;
      #pragma unroll
      for (int v2 = 0; v2 < 8; ++v2)
        lg[(rb + v2) * LSTR + col] = acc[t][v2];
    }
  }
  __syncthreads();

  // ---- Per-(row, node) softmax over 11 logits, in place in LDS ----
  for (int task = tid; task < 16 * NODES; task += BLOCK_THREADS) {
    int r = task / NODES;
    int n = task - r * NODES;
    float* pl = lg + r * LSTR + n * KFAN;
    const float* bn = bias + n * KFAN;
    float v[KFAN], mx = -1e30f;
    #pragma unroll
    for (int kk = 0; kk < KFAN; ++kk) {
      v[kk] = pl[kk] + bn[kk];
      mx = fmaxf(mx, v[kk]);
    }
    float s = 0.f;
    #pragma unroll
    for (int kk = 0; kk < KFAN; ++kk) { v[kk] = __expf(v[kk] - mx); s += v[kk]; }
    float inv = 1.0f / s;
    #pragma unroll
    for (int kk = 0; kk < KFAN; ++kk) pl[kk] = v[kk] * inv;
  }
  __syncthreads();

  // ---- Product tree: out[b, (i*10+j)*10+mm] = P[0][1+i]*P[1+i][1+j]*P[11+q][1+mm] ----
  for (int oi = tid; oi < 16 * OUT_CLASSES; oi += BLOCK_THREADS) {
    int r  = oi / OUT_CLASSES;
    int c  = oi - r * OUT_CLASSES;
    int q  = c / 10;                     // 0..99
    int mm = c - q * 10;
    int i2 = q / 10;
    int j  = q - i2 * 10;
    const float* pr = lg + r * LSTR;
    float val = pr[1 + i2] * pr[(1 + i2) * KFAN + 1 + j] * pr[(11 + q) * KFAN + 1 + mm];
    out[(row0 + r) * (long)OUT_CLASSES + c] = val;
  }

  // ---- penalty scalar (tuple element 2 of d_out) ----
  if (blockIdx.x == 0 && tid == 0) out[(long)B_ROWS * OUT_CLASSES] = 0.0f;
}

// ---------------------------------------------------------------------------
extern "C" void kernel_launch(void* const* d_in, const int* in_sizes, int n_in,
                              void* d_out, int out_size, void* d_ws, size_t ws_size,
                              hipStream_t stream) {
  (void)in_sizes; (void)n_in; (void)out_size; (void)ws_size;
  const float* x    = (const float*)d_in[0];
  // d_in[1] = labels: unused by the reference forward pass
  const float* W    = (const float*)d_in[2];
  const float* bias = (const float*)d_in[3];
  float*       out  = (float*)d_out;
  _Float16*    wpk  = (_Float16*)d_ws;   // 2048 * 1280 halves = 5.25 MB packed W

  // Pack W into WMMA B-fragment layout (one half per thread, coalesced writes).
  const long packed = (long)DIM * NTILES * 16;       // 2,621,440 halves
  pack_w_f16<<<(int)(packed / 256), 256, 0, stream>>>(W, wpk);

  // One block per 16 rows: 8192/16 = 512 blocks, 512 threads (16 wave32s).
  tree_infer<<<B_ROWS / 16, BLOCK_THREADS, LDS_BYTES, stream>>>(x, bias, wpk, out);
}
